// Interactive_Align_attention_8358006358493
// MI455X (gfx1250) — matrix operand order repaired and checked by reading, not compiled, not run on hardware
//
#include <hip/hip_runtime.h>
#include <hip/hip_bf16.h>

// ---------------------------------------------------------------------------
// Sizes fixed by the reference
// ---------------------------------------------------------------------------
#define B_   16
#define JX_  4096
#define JQ_  512
#define D_   256
#define CCK  1024           // 4*D  (cc feature dim)
#define VERY_NEG (-1e30f)

#define WPB  4              // waves per block (wave32)
#define SLAB_BYTES 49152    // 48 KB private slab per wave

// Block-shared staging buffers (filled by the Tensor Data Mover)
#define QT_OFF   (WPB * SLAB_BYTES)      // 196608 : qT tile  [16][512]  bf16 (16 KB)
#define WR_OFF   (QT_OFF + 16384)        // 212992 : WrT tile [16][1024] bf16 (32 KB)
#define WG_OFF   (WR_OFF + 32768)        // 245760 : WgT tile [16][1024] bf16 (32 KB)
#define LDS_TOTAL (WG_OFF + 32768)       // 278528 bytes  (<= 320 KB per WGP)

typedef __bf16 bf16;
typedef __attribute__((ext_vector_type(16))) __bf16 v16bf;
typedef __attribute__((ext_vector_type(8)))  float  v8f;
typedef __attribute__((ext_vector_type(4)))  float  f32x4;
typedef __attribute__((ext_vector_type(4)))  unsigned int u32x4;
typedef __attribute__((ext_vector_type(2)))  unsigned int u32x2;
typedef __attribute__((ext_vector_type(8)))  int    i32x8;
typedef __attribute__((ext_vector_type(4)))  int    i32x4;

union FragU { v16bf v; u32x4 q[2]; };

// ---------------------------------------------------------------------------
// Fragment loaders (CDNA5 WMMA 16x16x32 bf16 layouts, ISA 7.12.2)
//
// A (16x32, MxK): lane l<16 holds row M=l, K = {0..7, 16..23};
//                 lane l>=16 holds row M=l-16, K = {8..15, 24..31}.
// B (32x16, KxN): lane l holds col N=l&15, same K set as A for its lane half.
// C/D (16x16 f32): VGPR r, lanes 0-15 -> (M=r, N=lane); lanes 16-31 -> (M=8+r).
// ---------------------------------------------------------------------------
__device__ inline v16bf load_a_bf16(const bf16* base, int ld, int lane) {
  const int r = lane & 15, hi = lane >> 4;
  const bf16* p = base + r * ld + hi * 8;
  FragU u;
  u.q[0] = *(const u32x4*)(p);        // K span 0..7  (or 8..15)
  u.q[1] = *(const u32x4*)(p + 16);   // K span 16..23 (or 24..31)
  return u.v;
}

__device__ inline v16bf load_a_f32(const float* base, int ld, int lane) {
  const int r = lane & 15, hi = lane >> 4;
  const float* p = base + r * ld + hi * 8;
  f32x4 a = *(const f32x4*)(p);
  f32x4 b = *(const f32x4*)(p + 4);
  f32x4 c = *(const f32x4*)(p + 16);
  f32x4 d = *(const f32x4*)(p + 20);
  v16bf v;
  v[0]=(bf16)a.x;  v[1]=(bf16)a.y;  v[2]=(bf16)a.z;  v[3]=(bf16)a.w;
  v[4]=(bf16)b.x;  v[5]=(bf16)b.y;  v[6]=(bf16)b.z;  v[7]=(bf16)b.w;
  v[8]=(bf16)c.x;  v[9]=(bf16)c.y;  v[10]=(bf16)c.z; v[11]=(bf16)c.w;
  v[12]=(bf16)d.x; v[13]=(bf16)d.y; v[14]=(bf16)d.z; v[15]=(bf16)d.w;
  return v;
}

__device__ inline v16bf load_b_bf16(const bf16* base, int ld, int lane) {
  const int n = lane & 15, hi = lane >> 4;
  const bf16* p = base + n * ld + hi * 8;
  FragU u;
  u.q[0] = *(const u32x4*)(p);        // K span 0..7  (or 8..15)
  u.q[1] = *(const u32x4*)(p + 16);   // K span 16..23 (or 24..31)
  return u.v;
}

__device__ inline unsigned pack2bf(float a, float b) {
  union { bf16 h[2]; unsigned u; } x;
  x.h[0] = (bf16)a; x.h[1] = (bf16)b;
  return x.u;
}

// ---------------------------------------------------------------------------
// Tensor Data Mover: async 2-D bf16 tile load (global -> LDS).
// D# layout per CDNA5 ISA section 8.3/8.4:
//   group0: count=1, lds_addr, global_addr[56:0], type=2
//   group1: data_size=1 (2 bytes), tensor_dim0/1, tile_dim0/1, dim0 stride
// groups 2/3 zero (2-D tensor). Tracked by TENSORcnt.
// Toolchain here is the 6-arg clang-23 form:
//   (uint32x4 g0, int32x8 g1, int32x4 g2, int32x4 g3, int32x8 pad, i32 cpol)
// ---------------------------------------------------------------------------
__device__ inline void tdm_load_tile_2d(unsigned lds_off, const void* gptr,
                                        unsigned rows, unsigned row_elems,
                                        unsigned stride_elems) {
  const unsigned long long ga = (unsigned long long)(uintptr_t)gptr;
  u32x4 g0;
  g0[0] = 1u;                                         // count=1, user mode
  g0[1] = lds_off;                                    // lds_addr (bytes)
  g0[2] = (unsigned)(ga & 0xffffffffu);               // global_addr[31:0]
  g0[3] = (unsigned)((ga >> 32) & 0x01ffffffu)        // global_addr[56:32]
          | (2u << 30);                               // type = 2 ("image")
  i32x8 g1;
  g1[0] = (int)(1u << 16);                            // data_size = 1 -> 2 bytes
  g1[1] = (int)((row_elems & 0xffffu) << 16);         // tensor_dim0[15:0]
  g1[2] = (int)(((row_elems >> 16) & 0xffffu)         // tensor_dim0[31:16]
          | ((rows & 0xffffu) << 16));                // tensor_dim1[15:0]
  g1[3] = (int)(((rows >> 16) & 0xffffu)              // tensor_dim1[31:16]
          | ((row_elems & 0xffffu) << 16));           // tile_dim0
  g1[4] = (int)(rows & 0xffffu);                      // tile_dim1 (tile_dim2=0)
  g1[5] = (int)stride_elems;                          // tensor_dim0_stride[31:0]
  g1[6] = 0;                                          // stride hi / dim1_stride lo
  g1[7] = 0;
  i32x4 z4 = {0, 0, 0, 0};
  i32x8 z8 = {0, 0, 0, 0, 0, 0, 0, 0};
  __builtin_amdgcn_tensor_load_to_lds(g0, g1, z4, z4, z8, 0);
}

// ---------------------------------------------------------------------------
// Prep kernels: bf16 casts + transposes
// ---------------------------------------------------------------------------
__global__ void prep_q_kernel(const float* __restrict__ q,
                              bf16* __restrict__ q_bf,
                              bf16* __restrict__ qT_bf) {
  const int i = blockIdx.x * blockDim.x + threadIdx.x;   // 2^21 elements
  const int d = i & (D_ - 1);
  const int j = (i >> 8) & (JQ_ - 1);
  const int b = i >> 17;
  const float v = q[i];
  q_bf[i] = (bf16)v;
  qT_bf[((size_t)b * D_ + d) * JQ_ + j] = (bf16)v;
}

__global__ void prep_w_kernel(const float* __restrict__ Wr,
                              const float* __restrict__ Wg,
                              bf16* __restrict__ WrT,
                              bf16* __restrict__ WgT) {
  const int i = blockIdx.x * blockDim.x + threadIdx.x;   // 2^18 elements
  const int n = i & (D_ - 1);
  const int k = i >> 8;
  WrT[(size_t)n * CCK + k] = (bf16)Wr[i];
  WgT[(size_t)n * CCK + k] = (bf16)Wg[i];
}

// ---------------------------------------------------------------------------
// Fused kernel: one wave owns one (b, 16-row x-tile); 4 waves per block share
// one batch b and the TDM-staged B tiles.
// Per-wave 48 KB LDS slab:
//   phase 1: Pf f32 [16][512] @ [0,32K) ; Ph bf16 [16][512] @ [32K,48K)
//   phase 2: QA f32 [16][256] @ [0,16K)
//   phase 3: CC bf16 [16][1024] @ [16K,48K)
// ---------------------------------------------------------------------------
__global__ __launch_bounds__(WPB * 32)
void ia_fused_kernel(const float* __restrict__ c,
                     const bf16* __restrict__ q_bf,
                     const bf16* __restrict__ qT_bf,
                     const bf16* __restrict__ WrT,
                     const bf16* __restrict__ WgT,
                     const float* __restrict__ Br,
                     const float* __restrict__ Bg,
                     const unsigned char* __restrict__ c_mask,
                     const unsigned char* __restrict__ q_mask,
                     float* __restrict__ out) {
  extern __shared__ __align__(16) char lds_raw[];

  const int lane = threadIdx.x & 31;
  const int wv   = threadIdx.x >> 5;
  const int tile = blockIdx.x * WPB + wv;          // [0, 4096)
  const int b    = tile >> 8;                      // uniform per block
  const int x0   = (tile & 255) * 16;

  char* slab = lds_raw + (size_t)wv * SLAB_BYTES;
  float* Pf = (float*)slab;                        // [16][512] masked logits
  bf16*  Ph = (bf16*)(slab + 32768);               // [16][512] probs (bf16)
  float* QA = (float*)slab;                        // [16][256] q_a
  bf16*  CC = (bf16*)(slab + 16384);               // [16][1024] cc
  const bf16* QTl = (const bf16*)(lds_raw + QT_OFF);
  const bf16* WRl = (const bf16*)(lds_raw + WR_OFF);
  const bf16* WGl = (const bf16*)(lds_raw + WG_OFF);

  const int n  = lane & 15;
  const int hi = lane >> 4;

  // ---- A fragments of the c tile (convert f32->bf16 once, reuse 32x) ----
  v16bf aC[8];
  {
    const float* crow = c + ((size_t)b * JX_ + x0) * D_;
#pragma unroll
    for (int kc = 0; kc < 8; ++kc)
      aC[kc] = load_a_f32(crow + kc * 32, D_, lane);
  }

  // c_mask bits for the 8 rows this lane-half owns in the D fragment
  unsigned cm8 = 0;
#pragma unroll
  for (int r = 0; r < 8; ++r)
    cm8 |= (unsigned)(c_mask[(size_t)b * JX_ + x0 + hi * 8 + r] != 0) << r;

  // ---- GEMM1: logits = c @ q^T, masked, -> LDS (f32) ----
  for (int jt = 0; jt < 32; ++jt) {
    v8f acc = {};
    const bf16* qbase = q_bf + ((size_t)b * JQ_ + jt * 16) * D_;
#pragma unroll
    for (int kc = 0; kc < 8; ++kc) {
      v16bf bq = load_b_bf16(qbase + kc * 32, D_, lane);
      acc = __builtin_amdgcn_wmma_f32_16x16x32_bf16(
          false, aC[kc], false, bq, (short)0, acc, false, false);
    }
    const bool qm = q_mask[(size_t)b * JQ_ + jt * 16 + n] != 0;
#pragma unroll
    for (int r = 0; r < 8; ++r) {
      const bool cm = (cm8 >> r) & 1;
      Pf[(hi * 8 + r) * JQ_ + jt * 16 + n] =
          acc[r] + ((cm && qm) ? 0.0f : VERY_NEG);
    }
  }
  __syncthreads();

  // ---- softmax over j (each lane: row n, half hi -> 256 elements) ----
  {
    const f32x4* rp = (const f32x4*)(Pf + n * JQ_ + hi * 256);
    float mx = -3.0e38f;
    for (int j = 0; j < 64; ++j) {
      f32x4 v = rp[j];
      mx = fmaxf(mx, fmaxf(fmaxf(v.x, v.y), fmaxf(v.z, v.w)));
    }
    mx = fmaxf(mx, __shfl_xor(mx, 16, 32));
    float s = 0.0f;
    for (int j = 0; j < 64; ++j) {
      f32x4 v = rp[j];
      s += __expf(v.x - mx) + __expf(v.y - mx) + __expf(v.z - mx) + __expf(v.w - mx);
    }
    s += __shfl_xor(s, 16, 32);
    const float inv = 1.0f / s;
    bf16* prow = Ph + n * JQ_ + hi * 256;
    for (int j = 0; j < 64; ++j) {
      f32x4 v = rp[j];
      u32x2 w;
      w.x = pack2bf(__expf(v.x - mx) * inv, __expf(v.y - mx) * inv);
      w.y = pack2bf(__expf(v.z - mx) * inv, __expf(v.w - mx) * inv);
      *(u32x2*)(prow + j * 4) = w;
    }
  }
  __syncthreads();

  // ---- GEMM2: q_a = P @ q  (K=512, N=256); B tile TDM-staged per nt ----
  for (int nt = 0; nt < 16; ++nt) {
    if (wv == 0) {
      tdm_load_tile_2d(QT_OFF, qT_bf + ((size_t)b * D_ + nt * 16) * JQ_,
                       16, JQ_, JQ_);
      __builtin_amdgcn_s_wait_tensorcnt(0);
    }
    __syncthreads();
    v8f acc = {};
#pragma unroll
    for (int kc = 0; kc < 16; ++kc) {
      v16bf ap = load_a_bf16(Ph + kc * 32, JQ_, lane);
      v16bf bq = load_b_bf16(QTl + kc * 32, JQ_, lane);
      acc = __builtin_amdgcn_wmma_f32_16x16x32_bf16(
          false, ap, false, bq, (short)0, acc, false, false);
    }
#pragma unroll
    for (int r = 0; r < 8; ++r)
      QA[(hi * 8 + r) * D_ + nt * 16 + n] = acc[r];
    __syncthreads();
  }

  // ---- build cc tile [16][1024] bf16 in LDS ----
  {
    const float* crow  = c + ((size_t)b * JX_ + x0 + n) * D_;
    const float* qarow = QA + n * D_;
    bf16* ccrow = CC + n * CCK;
    for (int d0 = hi * 128; d0 < hi * 128 + 128; d0 += 4) {
      f32x4 cf = *(const f32x4*)(crow + d0);
      f32x4 qa = *(const f32x4*)(qarow + d0);
      u32x2 w;
      w.x = pack2bf(cf.x, cf.y);            w.y = pack2bf(cf.z, cf.w);
      *(u32x2*)(ccrow + d0) = w;                              // c
      w.x = pack2bf(qa.x, qa.y);            w.y = pack2bf(qa.z, qa.w);
      *(u32x2*)(ccrow + 256 + d0) = w;                        // q_a
      w.x = pack2bf(cf.x*qa.x, cf.y*qa.y);  w.y = pack2bf(cf.z*qa.z, cf.w*qa.w);
      *(u32x2*)(ccrow + 512 + d0) = w;                        // c*q_a
      w.x = pack2bf(cf.x-qa.x, cf.y-qa.y);  w.y = pack2bf(cf.z-qa.z, cf.w-qa.w);
      *(u32x2*)(ccrow + 768 + d0) = w;                        // c-q_a
    }
  }
  __syncthreads();

  // ---- GEMM3: r_ and g; W tiles TDM-staged per nt; fused epilogue ----
  for (int nt = 0; nt < 16; ++nt) {
    if (wv == 0) {
      tdm_load_tile_2d(WR_OFF, WrT + (size_t)(nt * 16) * CCK, 16, CCK, CCK);
      __builtin_amdgcn_s_wait_tensorcnt(0);
    } else if (wv == 1) {
      tdm_load_tile_2d(WG_OFF, WgT + (size_t)(nt * 16) * CCK, 16, CCK, CCK);
      __builtin_amdgcn_s_wait_tensorcnt(0);
    }
    __syncthreads();
    v8f ar = {};
    v8f ag = {};
#pragma unroll
    for (int kc = 0; kc < 32; ++kc) {
      v16bf a  = load_a_bf16(CC + kc * 32, CCK, lane);
      v16bf br = load_b_bf16(WRl + kc * 32, CCK, lane);
      v16bf bg = load_b_bf16(WGl + kc * 32, CCK, lane);
      ar = __builtin_amdgcn_wmma_f32_16x16x32_bf16(
          false, a, false, br, (short)0, ar, false, false);
      ag = __builtin_amdgcn_wmma_f32_16x16x32_bf16(
          false, a, false, bg, (short)0, ag, false, false);
    }
    const float brb = Br[nt * 16 + n];
    const float bgb = Bg[nt * 16 + n];
#pragma unroll
    for (int r = 0; r < 8; ++r) {
      const int m = hi * 8 + r;
      const size_t idx = ((size_t)b * JX_ + x0 + m) * D_ + nt * 16 + n;
      const float rv = tanhf(ar[r] + brb);
      const float gv = 1.0f / (1.0f + __expf(-(ag[r] + bgb)));
      const float cv = c[idx];
      out[idx] = gv * rv + (1.0f - gv) * cv;
    }
    __syncthreads();
  }
}

// ---------------------------------------------------------------------------
// Launch
// ---------------------------------------------------------------------------
extern "C" void kernel_launch(void* const* d_in, const int* in_sizes, int n_in,
                              void* d_out, int out_size, void* d_ws, size_t ws_size,
                              hipStream_t stream) {
  const float* c  = (const float*)d_in[0];
  const float* q  = (const float*)d_in[1];
  const float* Wr = (const float*)d_in[2];
  const float* Br = (const float*)d_in[3];
  const float* Wg = (const float*)d_in[4];
  const float* Bg = (const float*)d_in[5];
  const unsigned char* c_mask = (const unsigned char*)d_in[6];
  const unsigned char* q_mask = (const unsigned char*)d_in[7];
  float* out = (float*)d_out;

  // Workspace (bf16): q_bf 4 MiB | qT_bf 4 MiB | WrT 0.5 MiB | WgT 0.5 MiB
  char* ws = (char*)d_ws;
  const size_t qbytes = (size_t)B_ * JQ_ * D_ * sizeof(bf16);
  bf16* q_bf  = (bf16*)ws;
  bf16* qT_bf = (bf16*)(ws + qbytes);
  bf16* WrT   = (bf16*)(ws + 2 * qbytes);
  bf16* WgT   = WrT + (size_t)CCK * D_;

  prep_q_kernel<<<(B_ * JQ_ * D_) / 256, 256, 0, stream>>>(q, q_bf, qT_bf);
  prep_w_kernel<<<(CCK * D_) / 256, 256, 0, stream>>>(Wr, Wg, WrT, WgT);

  const int ntiles = B_ * JX_ / 16;                 // 4096 wave-tiles
  ia_fused_kernel<<<ntiles / WPB, WPB * 32, LDS_TOTAL, stream>>>(
      c, q_bf, qT_bf, WrT, WgT, Br, Bg, c_mask, q_mask, out);
}
